// CedrKnrmRanker_6305011990555
// MI455X (gfx1250) — matrix operand order repaired and checked by reading, not compiled, not run on hardware
//
#include <hip/hip_runtime.h>
#include <hip/hip_bf16.h>

// ---------------------------------------------------------------------------
// CEDR-KNRM ranker for MI455X (gfx1250, wave32).
// Memory-bound: 654MB of fp32 hidden states streamed once (~28us floor at
// 23.3 TB/s).  GEMM (20x492x768 per (l,b)) done with v_wmma_f32_16x16x32_bf16;
// normalization deferred to the epilogue (sim *= invq[m]*invd[n]) so each
// d-row is read from HBM exactly once.
// ---------------------------------------------------------------------------

typedef __attribute__((ext_vector_type(16))) __bf16 v16bf;
typedef __attribute__((ext_vector_type(8)))  __bf16 v8bf;
typedef __attribute__((ext_vector_type(4)))  __bf16 v4bf;
typedef __attribute__((ext_vector_type(4)))  float  v4f;
typedef __attribute__((ext_vector_type(8)))  float  v8f;

#define L_LAYERS 13
#define BATCH    32
#define SEQ      512
#define HDIM     768
#define QLEN     20
#define DLEN     492          // SEQ - QLEN
#define NK       11
#define NCHUNK   31           // ceil(492/16)
#define WAVES    8
#define THREADS  256

// LDS partition (bytes); total 245952 < 320KB WGP budget
#define Q_LDS_BYTES   (32 * HDIM * 2)          // 49152 : q rows 0..31 (20 real, 12 zero-pad)
#define D_LDS_BYTES   (WAVES * 16 * HDIM * 2)  // 196608: per-wave 16-row d tile
#define SMEM_TOTAL    (Q_LDS_BYTES + D_LDS_BYTES + 32 * 4 + 16 * 4)

static __device__ __forceinline__ __bf16 bf_zero() {
  return __builtin_bit_cast(__bf16, (unsigned short)0);
}

// float4 -> 4x bf16 via native packed converts (v_cvt_pk_bf16_f32)
static __device__ __forceinline__ v4bf pack4(float x, float y, float z, float w) {
  v4f t = {x, y, z, w};
  return __builtin_convertvector(t, v4bf);
}

__global__ __launch_bounds__(THREADS)
void knrm_sim_pool_kernel(const float* __restrict__ hs,
                          const float* __restrict__ mu,
                          const float* __restrict__ sigma,
                          float* __restrict__ pooled_out)
{
  extern __shared__ __align__(32) char smem[];
  __bf16* q_lds      = (__bf16*)smem;
  __bf16* d_lds_all  = (__bf16*)(smem + Q_LDS_BYTES);
  float*  invq       = (float*)(smem + Q_LDS_BYTES + D_LDS_BYTES);
  float*  pooled_lds = invq + 32;

  const int blk  = blockIdx.x;
  const int l    = blk >> 5;            // layer 0..12
  const int b    = blk & 31;            // batch
  const int tid  = threadIdx.x;
  const int wave = tid >> 5;
  const int lane = tid & 31;
  const int hf   = lane >> 4;           // lane half (0/1)
  const int l16  = lane & 15;

  // RBF kernel parameters per thread (uniform address -> scalar loads)
  float muv[NK], cf[NK];
#pragma unroll
  for (int k = 0; k < NK; ++k) {
    muv[k] = mu[k];
    float s = sigma[k];
    cf[k] = -0.5f / (s * s);
  }

  if (tid < 16) pooled_lds[tid] = 0.f;

  const size_t lb_base = (size_t)(l * BATCH + b) * SEQ * HDIM;

  // ---- Phase 1: stage q rows (tokens 0..19) as raw bf16 + fp32 inv-norms --
  for (int row = wave; row < QLEN; row += WAVES) {
    const float4* src = (const float4*)(hs + lb_base + (size_t)row * HDIM);
    float ss = 0.f;
#pragma unroll
    for (int i = 0; i < 6; ++i) {            // 6 iters * 32 lanes * float4 = 768
      float4 v = src[lane + 32 * i];
      ss += v.x * v.x + v.y * v.y + v.z * v.z + v.w * v.w;
      *(v4bf*)(&q_lds[row * HDIM + 4 * (lane + 32 * i)]) = pack4(v.x, v.y, v.z, v.w);
    }
#pragma unroll
    for (int off = 16; off > 0; off >>= 1) ss += __shfl_xor(ss, off, 32);
    if (lane == 0) invq[row] = 1.f / fmaxf(sqrtf(ss), 1e-8f);
  }
  // zero-pad q rows 20..31 so the second M-tile multiplies clean zeros
  for (int idx = tid; idx < 12 * HDIM; idx += THREADS)
    q_lds[QLEN * HDIM + idx] = bf_zero();
  __syncthreads();

  // ---- Phase 2: each wave streams its d chunks ----------------------------
  float pool[NK];
#pragma unroll
  for (int k = 0; k < NK; ++k) pool[k] = 0.f;

  auto pool_sim = [&](float sim) {
#pragma unroll
    for (int k = 0; k < NK; ++k) {
      float d = sim - muv[k];
      pool[k] += __expf(cf[k] * d * d);
    }
  };

  __bf16* d_lds = d_lds_all + wave * 16 * HDIM;

  for (int c = wave; c < NCHUNK; c += WAVES) {
    float invd_reg = 0.f;                     // inv-norm of row (lane&15)
    for (int r = 0; r < 16; ++r) {
      const int n = c * 16 + r;
      float ss = 0.f;
      if (n < DLEN) {                         // uniform per wave
        const float4* src = (const float4*)(hs + lb_base + (size_t)(QLEN + n) * HDIM);
#pragma unroll
        for (int i = 0; i < 6; ++i) {
          float4 v = src[lane + 32 * i];
          ss += v.x * v.x + v.y * v.y + v.z * v.z + v.w * v.w;
          *(v4bf*)(&d_lds[r * HDIM + 4 * (lane + 32 * i)]) = pack4(v.x, v.y, v.z, v.w);
        }
      } else {
        v4bf z = {bf_zero(), bf_zero(), bf_zero(), bf_zero()};
#pragma unroll
        for (int i = 0; i < 6; ++i)
          *(v4bf*)(&d_lds[r * HDIM + 4 * (lane + 32 * i)]) = z;
      }
#pragma unroll
      for (int off = 16; off > 0; off >>= 1) ss += __shfl_xor(ss, off, 32);
      float inv = (n < DLEN) ? (1.f / fmaxf(sqrtf(ss), 1e-8f)) : 0.f;
      if (l16 == r) invd_reg = inv;
    }
    __builtin_amdgcn_wave_barrier();          // wave-private LDS tile: DS in-order

    // raw-dot GEMM: sim_raw(32x16) = q_raw(32x768) * d_raw(768x16)
    v8f acc0 = {}; v8f acc1 = {};
    const int mrow0 = l16;                    // A layout: lane holds M = lane&15
    const int mrow1 = l16 + 16;
#pragma unroll 4
    for (int kb = 0; kb < HDIM; kb += 32) {
      // A frag (16x32 bf16): j0..15 -> K = kb+16*hf+j
      v8bf a0lo = *(const v8bf*)(&q_lds[mrow0 * HDIM + kb + 16 * hf]);
      v8bf a0hi = *(const v8bf*)(&q_lds[mrow0 * HDIM + kb + 16 * hf + 8]);
      v8bf a1lo = *(const v8bf*)(&q_lds[mrow1 * HDIM + kb + 16 * hf]);
      v8bf a1hi = *(const v8bf*)(&q_lds[mrow1 * HDIM + kb + 16 * hf + 8]);
      // B frag (32x16 bf16): lane holds column N = lane&15, K = kb+16*hf .. +15
      v16bf bfrag = *(const v16bf*)(&d_lds[l16 * HDIM + kb + 16 * hf]);
      v16bf a0, a1;
#pragma unroll
      for (int j = 0; j < 8; ++j) {
        a0[j] = a0lo[j]; a0[j + 8] = a0hi[j];
        a1[j] = a1lo[j]; a1[j + 8] = a1hi[j];
      }
      acc0 = __builtin_amdgcn_wmma_f32_16x16x32_bf16(false, a0, false, bfrag,
                                                     (short)0, acc0, false, false);
      acc1 = __builtin_amdgcn_wmma_f32_16x16x32_bf16(false, a1, false, bfrag,
                                                     (short)0, acc1, false, false);
    }

    // epilogue: normalize + RBF pooling.  C/D layout: N = lane&15, M = r + 8*hf.
    // Chunks 0..29 have all 16 columns valid -> fully branch-free path.
    if (c < NCHUNK - 1) {
#pragma unroll
      for (int r = 0; r < 8; ++r)
        pool_sim(acc0[r] * invq[r + 8 * hf] * invd_reg);
      if (hf == 0) {                          // tile1 rows 16..19 live in hf==0 lanes
#pragma unroll
        for (int r = 0; r < 4; ++r)
          pool_sim(acc1[r] * invq[r + 16] * invd_reg);
      }
    } else {
      const bool nvalid = (c * 16 + l16 < DLEN);
      if (nvalid) {
#pragma unroll
        for (int r = 0; r < 8; ++r)
          pool_sim(acc0[r] * invq[r + 8 * hf] * invd_reg);
      }
      if (nvalid && hf == 0) {
#pragma unroll
        for (int r = 0; r < 4; ++r)
          pool_sim(acc1[r] * invq[r + 16] * invd_reg);
      }
    }
  }

  __syncthreads();
#pragma unroll
  for (int k = 0; k < NK; ++k) atomicAdd(&pooled_lds[k], pool[k]);  // ds_add_f32
  __syncthreads();
  if (tid < NK) pooled_out[(size_t)blk * NK + tid] = pooled_lds[tid];
}

// ---- Final combine: out[b] = [cls(768) | feat(14*11)] . W^T + bias --------
__global__ __launch_bounds__(256)
void knrm_combine_kernel(const float* __restrict__ hs,
                         const float* __restrict__ W,
                         const float* __restrict__ bias,
                         const float* __restrict__ pooled,
                         float* __restrict__ out)
{
  const int b   = blockIdx.x;
  const int tid = threadIdx.x;
  float partial = 0.f;
  const float* cls = hs + (size_t)((L_LAYERS - 1) * BATCH + b) * SEQ * HDIM; // token 0
  for (int h = tid; h < HDIM; h += 256) partial += cls[h] * W[h];
  for (int i = tid; i < 14 * NK; i += 256) {
    const int lp = i / NK, k = i % NK;
    const int lsrc = (lp == 0) ? 0 : lp - 1;        // layer 0 duplicated
    partial += pooled[(size_t)(lsrc * BATCH + b) * NK + k] * W[HDIM + i];
  }
  __shared__ float red[8];
#pragma unroll
  for (int off = 16; off > 0; off >>= 1) partial += __shfl_xor(partial, off, 32);
  if ((tid & 31) == 0) red[tid >> 5] = partial;
  __syncthreads();
  if (tid == 0) {
    float t = 0.f;
#pragma unroll
    for (int w = 0; w < 8; ++w) t += red[w];
    out[b] = t + bias[0];
  }
}

extern "C" void kernel_launch(void* const* d_in, const int* in_sizes, int n_in,
                              void* d_out, int out_size, void* d_ws, size_t ws_size,
                              hipStream_t stream) {
  const float* hs    = (const float*)d_in[0];
  const float* mu    = (const float*)d_in[1];
  const float* sigma = (const float*)d_in[2];
  const float* W     = (const float*)d_in[3];
  const float* bias  = (const float*)d_in[4];
  float* out       = (float*)d_out;
  float* pooled_ws = (float*)d_ws;          // 13*32*11 floats

  (void)in_sizes; (void)n_in; (void)out_size; (void)ws_size;

  hipFuncSetAttribute((const void*)knrm_sim_pool_kernel,
                      hipFuncAttributeMaxDynamicSharedMemorySize, SMEM_TOTAL);

  knrm_sim_pool_kernel<<<L_LAYERS * BATCH, THREADS, SMEM_TOTAL, stream>>>(
      hs, mu, sigma, pooled_ws);
  knrm_combine_kernel<<<BATCH, 256, 0, stream>>>(hs, W, bias, pooled_ws, out);
}
